// Informer_14319420965098
// MI455X (gfx1250) — matrix operand order repaired and checked by reading, not compiled, not run on hardware
//
#include <hip/hip_runtime.h>
#include <math.h>

typedef __attribute__((ext_vector_type(16))) _Float16 v16h;
typedef __attribute__((ext_vector_type(8)))  _Float16 h8;
typedef __attribute__((ext_vector_type(8)))  float    v8f;

// ---------------------------------------------------------------------------
// WMMA GEMM: C[M,N] = act(A[M,K] @ W[K,N] + bias)
//  AMODE: 0 = A row-major (M,K); 1 = circular conv3 im2col of x(B,aL,aI), K=3*aI
//  BMODE: 0 = W row-major (K,N); 1 = conv weight OIL (N, bI, 3), K=3*bI
//  act:   0 = none, 1 = exact GELU
// Block 256 = 8 waves; tile 64x128; KC=32; 4 WMMA tiles per wave per K-step.
// A staged M-major (rows of 32 halves), B staged N-major so every fragment is
// built from contiguous 16B ds_load_b128 reads per ISA 7.12.2/7.12.4 layouts.
// ---------------------------------------------------------------------------
template <int AMODE, int BMODE>
__global__ __launch_bounds__(256) void gemm_wmma_t(
    const float* __restrict__ A, const float* __restrict__ Bw,
    const float* __restrict__ bias, float* __restrict__ C,
    int M, int N, int K, int aL, int aI, int bI, int act)
{
  __shared__ _Float16 Ah[64][32];   // [m][k]
  __shared__ _Float16 Bh[128][32];  // [n][k]  (transposed stage)
  const int tid  = threadIdx.x;
  const int m0   = blockIdx.y * 64;
  const int n0   = blockIdx.x * 128;
  const int wave = tid >> 5;
  const int lane = tid & 31;
  const int wr   = wave & 3;   // M subtile 0..3
  const int wc   = wave >> 2;  // N half: cols wc*64 .. wc*64+63
  v8f acc[4] = {{}, {}, {}, {}};

  for (int k0 = 0; k0 < K; k0 += 32) {
    // --- stage A 64x32 (f32 -> f16), one b128 store per thread ---
    {
      const int row = tid >> 2;
      const int kc0 = (tid & 3) * 8;
      const int gm  = m0 + row;
      h8 hv = {};
      if (gm < M) {
        if (AMODE == 0) {
          if (k0 + kc0 + 8 <= K) {
            const float4* p = (const float4*)(A + (size_t)gm * K + k0 + kc0);
            const float4 f0 = p[0], f1 = p[1];
            hv[0] = (_Float16)f0.x; hv[1] = (_Float16)f0.y;
            hv[2] = (_Float16)f0.z; hv[3] = (_Float16)f0.w;
            hv[4] = (_Float16)f1.x; hv[5] = (_Float16)f1.y;
            hv[6] = (_Float16)f1.z; hv[7] = (_Float16)f1.w;
          } else {
            #pragma unroll
            for (int j = 0; j < 8; ++j) {
              const int gk = k0 + kc0 + j;
              if (gk < K) hv[j] = (_Float16)A[(size_t)gm * K + gk];
            }
          }
        } else {
          const int b = gm / aL, l = gm % aL;
          #pragma unroll
          for (int j = 0; j < 8; ++j) {
            const int gk = k0 + kc0 + j;
            if (gk < K) {
              const int t = gk / aI, i = gk % aI;
              int ls = l + t - 1;
              ls = (ls < 0) ? ls + aL : (ls >= aL ? ls - aL : ls);
              hv[j] = (_Float16)A[((size_t)b * aL + ls) * aI + i];
            }
          }
        }
      }
      *(h8*)&Ah[row][kc0] = hv;
    }
    // --- stage B 32x128 into Bh[n][k]; coalesced global reads over n ---
    {
      const int kr  = tid >> 3;          // 0..31
      const int nc0 = (tid & 7) * 16;    // 0..112
      const int gk  = k0 + kr;
      const bool kin = gk < K;
      if (BMODE == 0) {
        #pragma unroll
        for (int j = 0; j < 16; ++j) {
          const int gn = n0 + nc0 + j;
          const float v = (kin && gn < N) ? Bw[(size_t)gk * N + gn] : 0.f;
          Bh[nc0 + j][kr] = (_Float16)v;
        }
      } else {
        const int t = kin ? gk / bI : 0, i = kin ? gk % bI : 0;
        #pragma unroll
        for (int j = 0; j < 16; ++j) {
          const int gn = n0 + nc0 + j;
          const float v = (kin && gn < N) ? Bw[((size_t)gn * bI + i) * 3 + t] : 0.f;
          Bh[nc0 + j][kr] = (_Float16)v;
        }
      }
    }
    __syncthreads();

    // --- fragments: all LDS reads are contiguous 16B (b128) ---
    const int mrow  = lane & 15;
    const int khalf = (lane >> 4) * 8;   // A: lanes 16-31 take the +8 K halves
    const int klo   = (lane >> 4) * 8;   // B: lanes 16-31 take the +8 K halves
    const int ncol  = lane & 15;
    v16h af = {};
    {
      const h8 a0 = *(const h8*)&Ah[wr * 16 + mrow][khalf];
      const h8 a1 = *(const h8*)&Ah[wr * 16 + mrow][16 + khalf];
      #pragma unroll
      for (int i = 0; i < 8; ++i) { af[i] = a0[i]; af[i + 8] = a1[i]; }
    }
    #pragma unroll
    for (int t = 0; t < 4; ++t) {
      const int n = wc * 64 + t * 16 + ncol;
      const h8 b0 = *(const h8*)&Bh[n][klo];
      const h8 b1 = *(const h8*)&Bh[n][klo + 16];
      v16h bf = {};
      #pragma unroll
      for (int i = 0; i < 8; ++i) { bf[i] = b0[i]; bf[i + 8] = b1[i]; }
      acc[t] = __builtin_amdgcn_wmma_f32_16x16x32_f16(false, af, false, bf,
                                                      (short)0, acc[t], false, false);
    }
    __syncthreads();
  }

  // --- epilogue: C layout VGPR v -> M = v + 8*(lane>=16), N = lane&15 ---
  const int ncol = lane & 15;
  const int mh   = (lane >> 4) * 8;
  #pragma unroll
  for (int v = 0; v < 8; ++v) {
    const int gm = m0 + wr * 16 + mh + v;
    if (gm >= M) continue;
    #pragma unroll
    for (int t = 0; t < 4; ++t) {
      const int gn = n0 + wc * 64 + t * 16 + ncol;
      if (gn >= N) continue;
      float val = acc[t][v];
      if (bias) val += bias[gn];
      if (act == 1) val = 0.5f * val * (1.0f + erff(val * 0.70710678118f));
      C[(size_t)gm * N + gn] = val;
    }
  }
}

// ---------------------------------------------------------------------------
// LayerNorm over D=512, optional fused residual: y = LN(x [+ res]) * g + b
// ---------------------------------------------------------------------------
__global__ __launch_bounds__(256) void ln512(
    const float* __restrict__ x, const float* __restrict__ res,
    const float* __restrict__ g, const float* __restrict__ bp,
    float* __restrict__ y)
{
  __shared__ float red[256];
  const size_t row = blockIdx.x;
  const int t = threadIdx.x;
  const float* xr = x + row * 512;
  float v0 = xr[t], v1 = xr[t + 256];
  if (res) { const float* rr = res + row * 512; v0 += rr[t]; v1 += rr[t + 256]; }
  red[t] = v0 + v1; __syncthreads();
  for (int o = 128; o > 0; o >>= 1) { if (t < o) red[t] += red[t + o]; __syncthreads(); }
  const float mean = red[0] * (1.0f / 512.0f);
  __syncthreads();
  const float d0 = v0 - mean, d1 = v1 - mean;
  red[t] = d0 * d0 + d1 * d1; __syncthreads();
  for (int o = 128; o > 0; o >>= 1) { if (t < o) red[t] += red[t + o]; __syncthreads(); }
  const float rstd = rsqrtf(red[0] * (1.0f / 512.0f) + 1e-5f);
  float* yr = y + row * 512;
  yr[t]       = d0 * rstd * g[t]       + bp[t];
  yr[t + 256] = d1 * rstd * g[t + 256] + bp[t + 256];
}

// ---------------------------------------------------------------------------
// ProbSparse sparsity measure + iterative top-u selection (one block per b,h)
// ---------------------------------------------------------------------------
__global__ __launch_bounds__(256) void prob_m_topk(
    const float* __restrict__ Q, const float* __restrict__ Kb,
    int Lq, int Lk, int u_part, int u, unsigned seed, int* __restrict__ top)
{
  __shared__ float m[736];
  __shared__ float rv[256];
  __shared__ int   ri[256];
  const int bh = blockIdx.x, b = bh >> 3, h = bh & 7;
  const int t = threadIdx.x;
  const float* qbase = Q  + (size_t)b * Lq * 512 + h * 64;
  const float* kbase = Kb + (size_t)b * Lk * 512 + h * 64;
  for (int l = t; l < Lq; l += 256) {
    const float* qr = qbase + (size_t)l * 512;
    float qv[64];
    #pragma unroll
    for (int d = 0; d < 64; ++d) qv[d] = qr[d];
    float mx = -3.4e38f, sm = 0.f;
    for (int s = 0; s < u_part; ++s) {
      unsigned hsh = (unsigned)l * 2654435761u ^ (unsigned)s * 0x9E3779B9u
                   ^ seed * 0x85EBCA6Bu;
      hsh ^= hsh >> 16; hsh *= 0x7FEB352Du; hsh ^= hsh >> 15;
      hsh *= 0x846CA68Bu; hsh ^= hsh >> 16;
      const int ks = (int)(hsh % (unsigned)Lk);
      const float* kr = kbase + (size_t)ks * 512;
      float dot = 0.f;
      #pragma unroll
      for (int d = 0; d < 64; ++d) dot += qv[d] * kr[d];
      mx = fmaxf(mx, dot); sm += dot;
    }
    m[l] = mx - sm / (float)Lk;
  }
  __syncthreads();
  for (int it = 0; it < u; ++it) {
    float best = -3.4e38f; int bi = 0;
    for (int l = t; l < Lq; l += 256)
      if (m[l] > best) { best = m[l]; bi = l; }
    rv[t] = best; ri[t] = bi; __syncthreads();
    for (int o = 128; o > 0; o >>= 1) {
      if (t < o && rv[t + o] > rv[t]) { rv[t] = rv[t + o]; ri[t] = ri[t + o]; }
      __syncthreads();
    }
    if (t == 0) { top[(size_t)bh * u + it] = ri[0]; m[ri[0]] = -3.4e38f; }
    __syncthreads();
  }
}

// ---------------------------------------------------------------------------
// Online-softmax attention for one query row (one wave per query).
// sel==nullptr -> dense (query = blockIdx.x); causal masks s > query index.
// Writes ctx[(bh*Lq + li)*64 + d].
// ---------------------------------------------------------------------------
__global__ __launch_bounds__(32) void attn_sel(
    const float* __restrict__ Q, const float* __restrict__ Kb,
    const float* __restrict__ Vb, float* __restrict__ ctx,
    const int* __restrict__ sel, int Lq, int Lk, int u, int causal)
{
  __shared__ float qs[64];
  __shared__ float accs[32][64];
  const int bh = blockIdx.y, b = bh >> 3, h = bh & 7;
  const int lane = threadIdx.x;
  const int li = sel ? sel[(size_t)bh * u + blockIdx.x] : (int)blockIdx.x;
  const float* qr = Q + ((size_t)b * Lq + li) * 512 + h * 64;
  qs[lane] = qr[lane]; qs[lane + 32] = qr[lane + 32];
  __syncthreads();
  const int smax = causal ? li : (Lk - 1);
  float lmax = -3.4e38f, lsum = 0.f;
  float acc[64];
  #pragma unroll
  for (int d = 0; d < 64; ++d) acc[d] = 0.f;
  const float* kbase = Kb + (size_t)b * Lk * 512 + h * 64;
  const float* vbase = Vb + (size_t)b * Lk * 512 + h * 64;
  for (int s = lane; s <= smax; s += 32) {
    const float* kr = kbase + (size_t)s * 512;
    float dot = 0.f;
    #pragma unroll
    for (int d = 0; d < 64; ++d) dot += qs[d] * kr[d];
    dot *= 0.125f;  // 1/sqrt(64)
    const float nm = fmaxf(lmax, dot);
    const float sc = __expf(lmax - nm);
    const float e  = __expf(dot - nm);
    const float* vr = vbase + (size_t)s * 512;
    lsum = lsum * sc + e;
    #pragma unroll
    for (int d = 0; d < 64; ++d) acc[d] = acc[d] * sc + e * vr[d];
    lmax = nm;
  }
  float gmax = lmax;
  #pragma unroll
  for (int o = 16; o > 0; o >>= 1) gmax = fmaxf(gmax, __shfl_xor(gmax, o, 32));
  const float cs = __expf(lmax - gmax);
  float gsum = lsum * cs;
  #pragma unroll
  for (int o = 16; o > 0; o >>= 1) gsum += __shfl_xor(gsum, o, 32);
  #pragma unroll
  for (int d = 0; d < 64; ++d) accs[lane][d] = acc[d] * cs;
  __syncthreads();
  const float inv = 1.0f / gsum;
  #pragma unroll
  for (int rep = 0; rep < 2; ++rep) {
    const int dd = lane + rep * 32;
    float s = 0.f;
    for (int r = 0; r < 32; ++r) s += accs[r][dd];
    ctx[((size_t)bh * Lq + li) * 64 + dd] = s * inv;
  }
}

// ---- context fills for unselected queries ----------------------------------
__global__ void ctx_mean_fill(const float* __restrict__ V, float* __restrict__ ctx,
                              int Lq, int Lk)
{
  const int bh = blockIdx.x, b = bh >> 3, h = bh & 7, d = threadIdx.x;  // 64 thr
  const float* vb = V + (size_t)b * Lk * 512 + h * 64 + d;
  float s = 0.f;
  for (int l = 0; l < Lk; ++l) s += vb[(size_t)l * 512];
  s /= (float)Lk;
  float* cb = ctx + (size_t)bh * Lq * 64 + d;
  for (int l = 0; l < Lq; ++l) cb[(size_t)l * 64] = s;
}

__global__ void ctx_cumsum_fill(const float* __restrict__ V, float* __restrict__ ctx,
                                int Lq)
{
  const int bh = blockIdx.x, b = bh >> 3, h = bh & 7, d = threadIdx.x;
  const float* vb = V + (size_t)b * Lq * 512 + h * 64 + d;
  float* cb = ctx + (size_t)bh * Lq * 64 + d;
  float run = 0.f;
  for (int l = 0; l < Lq; ++l) { run += vb[(size_t)l * 512]; cb[(size_t)l * 64] = run; }
}

// mix=False: out[b,l,h*64+d] = ctx[b,h,l,d]
__global__ void head_merge(const float* __restrict__ ctx, float* __restrict__ out, int L)
{
  const size_t idx = (size_t)blockIdx.x * 256 + threadIdx.x;
  const size_t n = (size_t)16 * L * 512;
  if (idx >= n) return;
  const int j = (int)(idx % 512);
  const size_t r = idx / 512;
  const int l = (int)(r % L), b = (int)(r / L);
  const int h = j >> 6, d = j & 63;
  out[idx] = ctx[(((size_t)(b * 8 + h) * L) + l) * 64 + d];
}

// ---- misc elementwise ------------------------------------------------------
__global__ void transpose_in(const float* __restrict__ hist, float* __restrict__ xe)
{
  const size_t idx = (size_t)blockIdx.x * 256 + threadIdx.x;
  const size_t n = (size_t)16 * 720 * 32;
  if (idx >= n) return;
  const int c = (int)(idx % 32);
  const size_t r = idx / 32;
  const int l = (int)(r % 720), b = (int)(r / 720);
  xe[idx] = hist[((size_t)b * 32 + c) * 720 + l];
}

__global__ void build_xdec(const float* __restrict__ xe, float* __restrict__ xd)
{
  const size_t idx = (size_t)blockIdx.x * 256 + threadIdx.x;
  const size_t n = (size_t)16 * 672 * 32;
  if (idx >= n) return;
  const int c = (int)(idx % 32);
  const size_t r = idx / 32;
  const int l = (int)(r % 672), b = (int)(r / 672);
  xd[idx] = (l < 336) ? xe[((size_t)b * 720 + 384 + l) * 32 + c] : 0.0f;
}

// marks: column-0 of emb, avg-pooled 4 levels (zero-padded 3-tap)
__global__ void compute_mark(const float* __restrict__ emb, float* __restrict__ mark,
                             int Lv)
{
  __shared__ float cur[768];
  const int t = threadIdx.x;
  if (t < Lv) cur[t] = emb[(size_t)t * 32];
  __syncthreads();
  for (int it = 0; it < 4; ++it) {
    if (t < Lv) mark[(size_t)t * 4 + it] = cur[t];
    __syncthreads();
    float nv = 0.f;
    if (t < Lv) {
      const float a = (t > 0) ? cur[t - 1] : 0.f;
      const float bb = cur[t];
      const float c = (t < Lv - 1) ? cur[t + 1] : 0.f;
      nv = (a + bb + c) * (1.0f / 3.0f);
    }
    __syncthreads();
    if (t < Lv) cur[t] = nv;
    __syncthreads();
  }
}

// x += positional(l,dd) + mark(l,:) @ temp_w(:,dd)
__global__ void embed_add(float* __restrict__ x, const float* __restrict__ mk_enc,
                          const float* __restrict__ mk_fut,
                          const float* __restrict__ temp_w, int L, int dec)
{
  const size_t idx = (size_t)blockIdx.x * 256 + threadIdx.x;
  const size_t n = (size_t)16 * L * 512;
  if (idx >= n) return;
  const int dd = (int)(idx % 512);
  const size_t r = idx / 512;
  const int l = (int)(r % L);
  const int pair = dd >> 1;
  const float freq = __expf(-(float)(2 * pair) * (9.21034037197618f / 512.0f));
  const float ang = (float)l * freq;
  const float pe = (dd & 1) ? __cosf(ang) : __sinf(ang);
  const float* mk;
  if (!dec) mk = mk_enc + (size_t)l * 4;
  else mk = (l < 336) ? mk_enc + (size_t)(384 + l) * 4 : mk_fut + (size_t)(l - 336) * 4;
  const float tv = mk[0] * temp_w[dd] + mk[1] * temp_w[512 + dd]
                 + mk[2] * temp_w[1024 + dd] + mk[3] * temp_w[1536 + dd];
  x[idx] += pe + tv;
}

// batchnorm over (B*L) rows per channel
__global__ __launch_bounds__(256) void bn_stats(const float* __restrict__ y,
                                                int rows, float* __restrict__ stats)
{
  __shared__ float s1[256], s2[256];
  const int c = blockIdx.x, t = threadIdx.x;
  float a = 0.f, b = 0.f;
  for (int r = t; r < rows; r += 256) {
    const float v = y[(size_t)r * 512 + c];
    a += v; b += v * v;
  }
  s1[t] = a; s2[t] = b; __syncthreads();
  for (int o = 128; o > 0; o >>= 1) {
    if (t < o) { s1[t] += s1[t + o]; s2[t] += s2[t + o]; }
    __syncthreads();
  }
  if (t == 0) {
    const float m = s1[0] / (float)rows;
    stats[c] = m;
    stats[512 + c] = s2[0] / (float)rows - m * m;
  }
}

__global__ void bn_apply_elu(float* __restrict__ y, const float* __restrict__ stats,
                             const float* __restrict__ g, const float* __restrict__ bb,
                             size_t n)
{
  const size_t idx = (size_t)blockIdx.x * 256 + threadIdx.x;
  if (idx >= n) return;
  const int c = (int)(idx % 512);
  const float z = (y[idx] - stats[c]) * rsqrtf(stats[512 + c] + 1e-5f) * g[c] + bb[c];
  y[idx] = z > 0.f ? z : (__expf(z) - 1.0f);
}

// maxpool window 3 stride 2 over time, -inf edge pad; out length L/2
__global__ void maxpool3s2(const float* __restrict__ y, float* __restrict__ out, int L)
{
  const int Lo = L / 2;
  const size_t idx = (size_t)blockIdx.x * 256 + threadIdx.x;
  const size_t n = (size_t)16 * Lo * 512;
  if (idx >= n) return;
  const int c = (int)(idx % 512);
  const size_t r = idx / 512;
  const int lo = (int)(r % Lo), b = (int)(r / Lo);
  const int li = 2 * lo;
  const float* base = y + (size_t)b * L * 512 + c;
  float mv = base[(size_t)li * 512];
  if (li - 1 >= 0) mv = fmaxf(mv, base[(size_t)(li - 1) * 512]);
  if (li + 1 < L)  mv = fmaxf(mv, base[(size_t)(li + 1) * 512]);
  out[idx] = mv;
}

// out (B,32,336) = fin(B,672,32)[:, -336:, :].transpose(0,2,1)
__global__ void slice_out(const float* __restrict__ fin, float* __restrict__ out)
{
  const size_t idx = (size_t)blockIdx.x * 256 + threadIdx.x;
  const size_t n = (size_t)16 * 32 * 336;
  if (idx >= n) return;
  const int l = (int)(idx % 336);
  const size_t r = idx / 336;
  const int c = (int)(r % 32), b = (int)(r / 32);
  out[idx] = fin[((size_t)b * 672 + 336 + l) * 32 + c];
}

// ---------------------------------------------------------------------------
extern "C" void kernel_launch(void* const* d_in, const int* in_sizes, int n_in,
                              void* d_out, int out_size, void* d_ws, size_t ws_size,
                              hipStream_t stream)
{
  (void)in_sizes; (void)n_in; (void)out_size; (void)ws_size;
  const int B = 16, H = 8, D = 512, DFF = 2048;
  const int SEQ = 720, OUTL = 336, LDEC = 672, CIN = 32, COUT = 32;

  const float* history = (const float*)d_in[0];
  int pi = 2;  // d_in[1] = future_data (unused by reference forward)
  auto nxt = [&]() { return (const float*)d_in[pi++]; };

  const float* enc_tok_w  = nxt();
  const float* enc_temp_w = nxt();
  const float* dec_tok_w  = nxt();
  const float* dec_temp_w = nxt();

  struct AttnP { const float *wq,*bq,*wk,*bk,*wv,*bv,*wo,*bo; };
  struct FfnP  { const float *w1,*b1,*w2,*b2; };
  struct LnP   { const float *g,*b; };
  auto getAttn = [&]() { AttnP a; a.wq=nxt(); a.bq=nxt(); a.wk=nxt(); a.bk=nxt();
                         a.wv=nxt(); a.bv=nxt(); a.wo=nxt(); a.bo=nxt(); return a; };
  auto getFfn  = [&]() { FfnP f; f.w1=nxt(); f.b1=nxt(); f.w2=nxt(); f.b2=nxt(); return f; };
  auto getLn   = [&]() { LnP l; l.g=nxt(); l.b=nxt(); return l; };

  AttnP eattn[3]; FfnP effn[3]; LnP eln1[3], eln2[3];
  for (int i = 0; i < 3; ++i) { eattn[i]=getAttn(); effn[i]=getFfn();
                                eln1[i]=getLn(); eln2[i]=getLn(); }
  const float *cw[2], *cb[2], *cbg[2], *cbb[2];
  for (int i = 0; i < 2; ++i) { cw[i]=nxt(); cb[i]=nxt(); cbg[i]=nxt(); cbb[i]=nxt(); }
  LnP enc_norm = getLn();
  AttnP dself[2], dcross[2]; FfnP dffn[2]; LnP dln1[2], dln2[2], dln3[2];
  for (int j = 0; j < 2; ++j) { dself[j]=getAttn(); dcross[j]=getAttn();
                                dffn[j]=getFfn(); dln1[j]=getLn();
                                dln2[j]=getLn(); dln3[j]=getLn(); }
  LnP dec_norm = getLn();
  const float* proj_w  = nxt();
  const float* proj_b  = nxt();
  const float* hist_emb = nxt();
  const float* fut_emb  = nxt();

  // ---- workspace bump allocator ----
  size_t off = 0;
  auto alloc = [&](size_t nfloat) -> float* {
    float* p = (float*)((char*)d_ws + off);
    off += (nfloat * sizeof(float) + 255) & ~(size_t)255;
    return p;
  };
  float* x_enc   = alloc((size_t)B * SEQ * CIN);
  float* x_decin = alloc((size_t)B * LDEC * CIN);
  float* mk_enc  = alloc((size_t)SEQ * 4);
  float* mk_fut  = alloc((size_t)OUTL * 4);
  const size_t ACT = (size_t)B * SEQ * D;
  float* X   = alloc(ACT);
  float* X2  = alloc(ACT);
  float* Qb  = alloc(ACT);
  float* Kb  = alloc(ACT);
  float* Vb  = alloc(ACT);
  float* CTX = alloc(ACT);
  float* AO  = alloc(ACT);
  float* TMP = alloc((size_t)B * SEQ * DFF);
  int*   TOPI = (int*)alloc(8192);
  float* BNS = alloc(1024);
  float* ENC = alloc((size_t)B * 180 * D);
  float* FIN = alloc((size_t)B * LDEC * COUT);

  auto eg = [](size_t n) { return dim3((unsigned)((n + 255) / 256)); };
  // mode 0: linear A & W; mode 1: circular-conv3 A + OIL conv weights
  auto gemm = [&](const float* A, const float* Bw, const float* bias, float* Cc,
                  int M, int N, int Kd, int mode, int aL, int aI, int bI, int act) {
    dim3 g((N + 127) / 128, (M + 63) / 64);
    if (mode == 0)
      gemm_wmma_t<0, 0><<<g, 256, 0, stream>>>(A, Bw, bias, Cc, M, N, Kd, 0, 0, 0, act);
    else
      gemm_wmma_t<1, 1><<<g, 256, 0, stream>>>(A, Bw, bias, Cc, M, N, Kd, aL, aI, bI, act);
  };
  auto uval = [](int L) { int v = 5 * (int)ceil(log((double)L)); return v < L ? v : L; };

  // ---- input prep ----
  transpose_in<<<eg((size_t)B*SEQ*CIN), 256, 0, stream>>>(history, x_enc);
  build_xdec<<<eg((size_t)B*LDEC*CIN), 256, 0, stream>>>(x_enc, x_decin);
  compute_mark<<<1, 768, 0, stream>>>(hist_emb, mk_enc, SEQ);
  compute_mark<<<1, 768, 0, stream>>>(fut_emb, mk_fut, OUTL);

  // ---- encoder embedding (circular conv3 as WMMA GEMM, K=96) ----
  gemm(x_enc, enc_tok_w, nullptr, X, B*SEQ, D, 3*CIN, 1, SEQ, CIN, CIN, 0);
  embed_add<<<eg((size_t)B*SEQ*D), 256, 0, stream>>>(X, mk_enc, mk_fut, enc_temp_w, SEQ, 0);

  // ---- encoder layers ----
  int L = SEQ;
  for (int i = 0; i < 3; ++i) {
    const int M = B * L;
    gemm(X, eattn[i].wq, eattn[i].bq, Qb, M, D, D, 0, 0, 0, 0, 0);
    gemm(X, eattn[i].wk, eattn[i].bk, Kb, M, D, D, 0, 0, 0, 0, 0);
    gemm(X, eattn[i].wv, eattn[i].bv, Vb, M, D, D, 0, 0, 0, 0, 0);
    const int u = uval(L);
    prob_m_topk<<<B*H, 256, 0, stream>>>(Qb, Kb, L, L, u, u, (unsigned)(i+1), TOPI);
    ctx_mean_fill<<<B*H, 64, 0, stream>>>(Vb, CTX, L, L);
    attn_sel<<<dim3(u, B*H), 32, 0, stream>>>(Qb, Kb, Vb, CTX, TOPI, L, L, u, 0);
    head_merge<<<eg((size_t)M*D), 256, 0, stream>>>(CTX, TMP, L);
    gemm(TMP, eattn[i].wo, eattn[i].bo, AO, M, D, D, 0, 0, 0, 0, 0);
    ln512<<<M, 256, 0, stream>>>(X, AO, eln1[i].g, eln1[i].b, X2);
    gemm(X2, effn[i].w1, effn[i].b1, TMP, M, DFF, D, 0, 0, 0, 0, 1);
    gemm(TMP, effn[i].w2, effn[i].b2, AO, M, D, DFF, 0, 0, 0, 0, 0);
    ln512<<<M, 256, 0, stream>>>(X2, AO, eln2[i].g, eln2[i].b, X);
    if (i < 2) {  // distill: circular conv (K=1536) + BN + ELU + maxpool s2
      gemm(X, cw[i], cb[i], X2, M, D, 3*D, 1, L, D, D, 0);
      bn_stats<<<D, 256, 0, stream>>>(X2, M, BNS);
      bn_apply_elu<<<eg((size_t)M*D), 256, 0, stream>>>(X2, BNS, cbg[i], cbb[i], (size_t)M*D);
      maxpool3s2<<<eg((size_t)B*(L/2)*D), 256, 0, stream>>>(X2, X, L);
      L /= 2;
    }
  }
  ln512<<<B*L, 256, 0, stream>>>(X, nullptr, enc_norm.g, enc_norm.b, ENC);
  const int LE = L;  // 180

  // ---- decoder embedding ----
  gemm(x_decin, dec_tok_w, nullptr, X, B*LDEC, D, 3*CIN, 1, LDEC, CIN, CIN, 0);
  embed_add<<<eg((size_t)B*LDEC*D), 256, 0, stream>>>(X, mk_enc, mk_fut, dec_temp_w, LDEC, 1);

  // ---- decoder layers ----
  for (int j = 0; j < 2; ++j) {
    const int M = B * LDEC;
    // masked ProbSparse self-attention (mix=True: ctx flat feeds wo directly)
    gemm(X, dself[j].wq, dself[j].bq, Qb, M, D, D, 0, 0, 0, 0, 0);
    gemm(X, dself[j].wk, dself[j].bk, Kb, M, D, D, 0, 0, 0, 0, 0);
    gemm(X, dself[j].wv, dself[j].bv, Vb, M, D, D, 0, 0, 0, 0, 0);
    const int u = uval(LDEC);
    prob_m_topk<<<B*H, 256, 0, stream>>>(Qb, Kb, LDEC, LDEC, u, u, (unsigned)(1000+j), TOPI);
    ctx_cumsum_fill<<<B*H, 64, 0, stream>>>(Vb, CTX, LDEC);
    attn_sel<<<dim3(u, B*H), 32, 0, stream>>>(Qb, Kb, Vb, CTX, TOPI, LDEC, LDEC, u, 1);
    gemm(CTX, dself[j].wo, dself[j].bo, AO, M, D, D, 0, 0, 0, 0, 0);
    ln512<<<M, 256, 0, stream>>>(X, AO, dln1[j].g, dln1[j].b, X2);
    // full cross-attention vs encoder output
    gemm(X2, dcross[j].wq, dcross[j].bq, Qb, M, D, D, 0, 0, 0, 0, 0);
    gemm(ENC, dcross[j].wk, dcross[j].bk, Kb, B*LE, D, D, 0, 0, 0, 0, 0);
    gemm(ENC, dcross[j].wv, dcross[j].bv, Vb, B*LE, D, D, 0, 0, 0, 0, 0);
    attn_sel<<<dim3(LDEC, B*H), 32, 0, stream>>>(Qb, Kb, Vb, CTX, nullptr, LDEC, LE, 0, 0);
    head_merge<<<eg((size_t)M*D), 256, 0, stream>>>(CTX, TMP, LDEC);
    gemm(TMP, dcross[j].wo, dcross[j].bo, AO, M, D, D, 0, 0, 0, 0, 0);
    ln512<<<M, 256, 0, stream>>>(X2, AO, dln2[j].g, dln2[j].b, X);
    // FFN
    gemm(X, dffn[j].w1, dffn[j].b1, TMP, M, DFF, D, 0, 0, 0, 0, 1);
    gemm(TMP, dffn[j].w2, dffn[j].b2, AO, M, D, DFF, 0, 0, 0, 0, 0);
    ln512<<<M, 256, 0, stream>>>(X, AO, dln3[j].g, dln3[j].b, X2);
    float* t = X; X = X2; X2 = t;
  }
  ln512<<<B*LDEC, 256, 0, stream>>>(X, nullptr, dec_norm.g, dec_norm.b, X2);
  gemm(X2, proj_w, proj_b, FIN, B*LDEC, COUT, D, 0, 0, 0, 0, 0);
  slice_out<<<eg((size_t)B*COUT*OUTL), 256, 0, stream>>>(FIN, (float*)d_out);
}
